// FP_Layer_59717225284056
// MI455X (gfx1250) — compile-verified, matrix operand order checked by reading, not run on hardware
//
#include <hip/hip_runtime.h>
#include <hip/hip_bf16.h>
#include <math.h>
#include <stdint.h>

// Problem constants (from reference)
#define B_    8
#define N1_   8192
#define N2_   2048
#define C1_   128
#define C2_   256
#define INCH_ 384          // C2 + C1
#define O1_   256
#define O2_   256

typedef __attribute__((ext_vector_type(16))) __bf16 v16bf;
typedef __attribute__((ext_vector_type(8)))  float  v8f;

// ---------------------------------------------------------------------------
// Fragment loader for 16-bit WMMA A/B operands (CDNA5 ISA 7.12.2 layout).
// For lane l: row/col = l & 15, kbase = (l >= 16) ? 8 : 0.
// VGPR i holds K = ((i>>2)*16) + kbase + 2*(i&3) + {0,1}  -> one aligned dword
// per VGPR when the source row is K-contiguous bf16.
// ---------------------------------------------------------------------------
__device__ __forceinline__ v16bf load_frag16(const __bf16* __restrict__ p, int kb) {
    union { v16bf v; unsigned u[8]; } r;
    const unsigned* p32 = (const unsigned*)p;   // rows are 4B-aligned (even strides)
#pragma unroll
    for (int i = 0; i < 8; ++i) {
        int k = ((i >> 2) << 4) + kb + ((i & 3) << 1);
        r.u[i] = p32[k >> 1];
    }
    return r.v;
}

// ---------------------------------------------------------------------------
// Kernel 1: fold BN scale into weights (bf16) and bias (f32).
//   s = g / sqrt(1 + eps);  W' = s*W;  b' = s*b + beta
// ---------------------------------------------------------------------------
__global__ void prep_weights(const float* __restrict__ W1, const float* __restrict__ b1,
                             const float* __restrict__ g1, const float* __restrict__ be1,
                             const float* __restrict__ W2, const float* __restrict__ b2,
                             const float* __restrict__ g2, const float* __restrict__ be2,
                             __bf16* __restrict__ W1q, __bf16* __restrict__ W2q,
                             float* __restrict__ bias1, float* __restrict__ bias2) {
    const float s = 1.0f / sqrtf(1.0f + 1e-5f);
    int t = blockIdx.x * blockDim.x + threadIdx.x;
    int stride = gridDim.x * blockDim.x;
    for (int i = t; i < O1_ * INCH_; i += stride) {
        int o = i / INCH_;
        W1q[i] = (__bf16)(g1[o] * s * W1[i]);
    }
    for (int i = t; i < O2_ * O1_; i += stride) {
        int o = i / O1_;
        W2q[i] = (__bf16)(g2[o] * s * W2[i]);
    }
    for (int i = t; i < O1_; i += stride) bias1[i] = g1[i] * s * b1[i] + be1[i];
    for (int i = t; i < O2_; i += stride) bias2[i] = g2[i] * s * b2[i] + be2[i];
}

// ---------------------------------------------------------------------------
// Kernel 2: transpose feats2 (B, C2, N2) f32 -> (B, N2, C2) bf16 so the
// interpolation gathers become contiguous 512B rows instead of 4B scatters.
// Classic 32x32 LDS tile transpose, 256 threads.
// ---------------------------------------------------------------------------
__global__ void transpose_f2(const float* __restrict__ f2, __bf16* __restrict__ f2t) {
    __shared__ float tile[32][33];
    const int b  = blockIdx.z;
    const int c0 = blockIdx.y * 32;
    const int n0 = blockIdx.x * 32;
    const int tx = threadIdx.x & 31;
    const int ty = threadIdx.x >> 5;   // 0..7
#pragma unroll
    for (int i = ty; i < 32; i += 8)
        tile[i][tx] = f2[((size_t)b * C2_ + c0 + i) * N2_ + n0 + tx];
    __syncthreads();
#pragma unroll
    for (int i = ty; i < 32; i += 8)
        f2t[((size_t)b * N2_ + n0 + i) * C2_ + c0 + tx] = (__bf16)tile[tx][i];
}

// ---------------------------------------------------------------------------
// Kernel 3: 3-NN inverse-distance interpolation + concat.
// One thread per (b, n1) point. xyz2 for the batch (24 KB) staged in LDS via
// gfx1250 async-to-LDS loads (ASYNCcnt tracked), then top-3 scan.
// Output X: (B, N1, INCH) bf16, channel-contiguous rows (WMMA-B friendly).
// ---------------------------------------------------------------------------
__global__ void knn_interp(const float* __restrict__ xyz1, const float* __restrict__ xyz2,
                           const float* __restrict__ f1,   const __bf16* __restrict__ f2t,
                           __bf16* __restrict__ X) {
    __shared__ float s2[N2_ * 3];      // 24576 B
    const int b = blockIdx.y;
    const int n = blockIdx.x * blockDim.x + threadIdx.x;

    // Async copy: 1536 x 16B chunks over 256 threads (6 per thread).
    {
        unsigned lbase = (unsigned)(uintptr_t)(void*)&s2[0];
        unsigned long long gbase =
            (unsigned long long)(uintptr_t)(xyz2 + (size_t)b * N2_ * 3);
#pragma unroll
        for (int i = 0; i < 6; ++i) {
            unsigned           loff = lbase + (unsigned)(i * 256 + threadIdx.x) * 16u;
            unsigned long long ga   = gbase + (unsigned long long)(i * 256 + threadIdx.x) * 16u;
            asm volatile("global_load_async_to_lds_b128 %0, %1, off"
                         :: "v"(loff), "v"(ga) : "memory");
        }
        asm volatile("s_wait_asynccnt 0" ::: "memory");
    }
    __syncthreads();

    const float px = xyz1[((size_t)b * N1_ + n) * 3 + 0];
    const float py = xyz1[((size_t)b * N1_ + n) * 3 + 1];
    const float pz = xyz1[((size_t)b * N1_ + n) * 3 + 2];

    float d0 = 3.4e38f, d1 = 3.4e38f, d2 = 3.4e38f;
    int   i0 = 0,       i1 = 0,       i2 = 0;
    for (int j = 0; j < N2_; ++j) {
        float dx = px - s2[j * 3 + 0];
        float dy = py - s2[j * 3 + 1];
        float dz = pz - s2[j * 3 + 2];
        float dd = dx * dx + dy * dy + dz * dz;
        if (dd < d0)      { d2 = d1; i2 = i1; d1 = d0; i1 = i0; d0 = dd; i0 = j; }
        else if (dd < d1) { d2 = d1; i2 = i1; d1 = dd; i1 = j; }
        else if (dd < d2) { d2 = dd; i2 = j; }
    }
    // weights: w_k = 1/max(sqrt(d2_k), 1e-8), normalized
    float w0 = 1.0f / fmaxf(sqrtf(fmaxf(d0, 0.0f)), 1e-8f);
    float w1 = 1.0f / fmaxf(sqrtf(fmaxf(d1, 0.0f)), 1e-8f);
    float w2 = 1.0f / fmaxf(sqrtf(fmaxf(d2, 0.0f)), 1e-8f);
    float wn = 1.0f / (w0 + w1 + w2);
    w0 *= wn; w1 *= wn; w2 *= wn;

    __bf16* __restrict__ row = X + ((size_t)b * N1_ + n) * INCH_;
    const __bf16* __restrict__ r0p = f2t + ((size_t)b * N2_ + i0) * C2_;
    const __bf16* __restrict__ r1p = f2t + ((size_t)b * N2_ + i1) * C2_;
    const __bf16* __restrict__ r2p = f2t + ((size_t)b * N2_ + i2) * C2_;
#pragma unroll 8
    for (int c = 0; c < C2_; ++c) {
        float v = w0 * (float)r0p[c] + w1 * (float)r1p[c] + w2 * (float)r2p[c];
        row[c] = (__bf16)v;
    }
    const float* __restrict__ f1b = f1 + (size_t)b * C1_ * N1_ + n;
#pragma unroll 4
    for (int c = 0; c < C1_; ++c)
        row[C2_ + c] = (__bf16)f1b[(size_t)c * N1_];
}

// ---------------------------------------------------------------------------
// Kernel 4: layer 1 GEMM  H1 = relu(W1' @ X + b1').
// Per wave: 32(M) x 64(N) tile, K = 384 -> 12 k-steps, 8 WMMAs each
// (each B fragment reused by 2 A fragments: 12 b128 loads / 8 WMMAs).
// Output H1: (B, N1, O1) bf16, channel-contiguous rows.
// ---------------------------------------------------------------------------
__global__ void gemm_layer1(const __bf16* __restrict__ W, const float* __restrict__ bias,
                            const __bf16* __restrict__ X, __bf16* __restrict__ H) {
    const int lane = threadIdx.x & 31;
    const int gw   = blockIdx.x * (blockDim.x >> 5) + (threadIdx.x >> 5);
    const int NB   = N1_ / 64;                 // 128
    const int MT   = O1_ / 32;                 // 8
    const int nb   = gw % NB;
    const int mt   = (gw / NB) % MT;
    const int b    = gw / (NB * MT);
    const int m0   = mt * 32;
    const int n0   = nb * 64;
    const int kb   = ((lane >> 4) & 1) * 8;
    const int lid  = lane & 15;

    const __bf16* __restrict__ A0 = W + (size_t)(m0 + lid) * INCH_;
    const __bf16* __restrict__ A1 = W + (size_t)(m0 + 16 + lid) * INCH_;
    const __bf16* __restrict__ Xb = X + ((size_t)b * N1_ + n0) * INCH_;

    v8f acc[8] = {};
#pragma unroll 2
    for (int ks = 0; ks < INCH_; ks += 32) {
        v16bf a0 = load_frag16(A0 + ks, kb);
        v16bf a1 = load_frag16(A1 + ks, kb);
        __builtin_prefetch(Xb + (size_t)lid * INCH_ + ks + 64, 0, 1);
#pragma unroll
        for (int t = 0; t < 4; ++t) {
            v16bf bf = load_frag16(Xb + (size_t)(t * 16 + lid) * INCH_ + ks, kb);
            acc[t]     = __builtin_amdgcn_wmma_f32_16x16x32_bf16(
                false, a0, false, bf, (short)0, acc[t],     false, false);
            acc[4 + t] = __builtin_amdgcn_wmma_f32_16x16x32_bf16(
                false, a1, false, bf, (short)0, acc[4 + t], false, false);
        }
    }

#pragma unroll
    for (int h = 0; h < 2; ++h) {
        const int mb = m0 + 16 * h + ((lane >> 4) & 1) * 8;
#pragma unroll
        for (int t = 0; t < 4; ++t) {
            __bf16* __restrict__ dst =
                H + ((size_t)b * N1_ + n0 + t * 16 + lid) * O1_ + mb;
#pragma unroll
            for (int r = 0; r < 8; ++r) {
                float v = fmaxf(acc[4 * h + t][r] + bias[mb + r], 0.0f);
                dst[r] = (__bf16)v;
            }
        }
    }
}

// ---------------------------------------------------------------------------
// Kernel 5: layer 2 GEMM  out = relu(W2' @ H1 + b2'), K = 256 -> 8 k-steps.
// Per wave: 32(M) x 64(N). Output: (B, O2, N1) f32 straight into d_out.
// ---------------------------------------------------------------------------
__global__ void gemm_layer2(const __bf16* __restrict__ W, const float* __restrict__ bias,
                            const __bf16* __restrict__ H, float* __restrict__ out) {
    const int lane = threadIdx.x & 31;
    const int gw   = blockIdx.x * (blockDim.x >> 5) + (threadIdx.x >> 5);
    const int NB   = N1_ / 64;
    const int MT   = O2_ / 32;
    const int nb   = gw % NB;
    const int mt   = (gw / NB) % MT;
    const int b    = gw / (NB * MT);
    const int m0   = mt * 32;
    const int n0   = nb * 64;
    const int kb   = ((lane >> 4) & 1) * 8;
    const int lid  = lane & 15;

    const __bf16* __restrict__ A0 = W + (size_t)(m0 + lid) * O1_;
    const __bf16* __restrict__ A1 = W + (size_t)(m0 + 16 + lid) * O1_;
    const __bf16* __restrict__ Hb = H + ((size_t)b * N1_ + n0) * O1_;

    v8f acc[8] = {};
#pragma unroll 2
    for (int ks = 0; ks < O1_; ks += 32) {
        v16bf a0 = load_frag16(A0 + ks, kb);
        v16bf a1 = load_frag16(A1 + ks, kb);
        __builtin_prefetch(Hb + (size_t)lid * O1_ + ks + 64, 0, 1);
#pragma unroll
        for (int t = 0; t < 4; ++t) {
            v16bf bf = load_frag16(Hb + (size_t)(t * 16 + lid) * O1_ + ks, kb);
            acc[t]     = __builtin_amdgcn_wmma_f32_16x16x32_bf16(
                false, a0, false, bf, (short)0, acc[t],     false, false);
            acc[4 + t] = __builtin_amdgcn_wmma_f32_16x16x32_bf16(
                false, a1, false, bf, (short)0, acc[4 + t], false, false);
        }
    }

    float* __restrict__ ob = out + (size_t)b * O2_ * N1_;
#pragma unroll
    for (int h = 0; h < 2; ++h) {
        const int mb = m0 + 16 * h + ((lane >> 4) & 1) * 8;
#pragma unroll
        for (int t = 0; t < 4; ++t) {
            const int nn = n0 + t * 16 + lid;
#pragma unroll
            for (int r = 0; r < 8; ++r) {
                float v = fmaxf(acc[4 * h + t][r] + bias[mb + r], 0.0f);
                ob[(size_t)(mb + r) * N1_ + nn] = v;
            }
        }
    }
}

// ---------------------------------------------------------------------------
// Host launcher
// ---------------------------------------------------------------------------
extern "C" void kernel_launch(void* const* d_in, const int* in_sizes, int n_in,
                              void* d_out, int out_size, void* d_ws, size_t ws_size,
                              hipStream_t stream) {
    const float* xyz1 = (const float*)d_in[0];
    const float* xyz2 = (const float*)d_in[1];
    const float* f1   = (const float*)d_in[2];
    const float* f2   = (const float*)d_in[3];
    const float* W1   = (const float*)d_in[4];
    const float* b1   = (const float*)d_in[5];
    const float* g1   = (const float*)d_in[6];
    const float* be1  = (const float*)d_in[7];
    const float* W2   = (const float*)d_in[8];
    const float* b2   = (const float*)d_in[9];
    const float* g2   = (const float*)d_in[10];
    const float* be2  = (const float*)d_in[11];
    float* out = (float*)d_out;

    // Workspace carving (all offsets 256B-aligned)
    char* ws = (char*)d_ws;
    size_t off = 0;
    __bf16* W1q = (__bf16*)(ws + off); off += (size_t)O1_ * INCH_ * 2;          // 192 KB
    __bf16* W2q = (__bf16*)(ws + off); off += (size_t)O2_ * O1_ * 2;            // 128 KB
    float* bias1 = (float*)(ws + off); off += O1_ * 4;
    float* bias2 = (float*)(ws + off); off += O2_ * 4;
    off = (off + 255) & ~(size_t)255;
    __bf16* f2t = (__bf16*)(ws + off); off += (size_t)B_ * N2_ * C2_ * 2;       // 8 MB
    __bf16* X   = (__bf16*)(ws + off); off += (size_t)B_ * N1_ * INCH_ * 2;     // 48 MB
    __bf16* H1  = (__bf16*)(ws + off); off += (size_t)B_ * N1_ * O1_ * 2;       // 32 MB

    // 1) fold BN into weights/bias, convert to bf16
    prep_weights<<<64, 256, 0, stream>>>(W1, b1, g1, be1, W2, b2, g2, be2,
                                         W1q, W2q, bias1, bias2);

    // 2) feats2 (B,C2,N2) f32 -> (B,N2,C2) bf16 for contiguous neighbor rows
    transpose_f2<<<dim3(N2_ / 32, C2_ / 32, B_), 256, 0, stream>>>(f2, f2t);

    // 3) KNN + inverse-distance interpolation + concat -> X (B, N1, 384) bf16
    knn_interp<<<dim3(N1_ / 256, B_), 256, 0, stream>>>(xyz1, xyz2, f1, f2t, X);

    // 4) MLP layer 1: waves = B*(O1/32)*(N1/64) = 8192 -> 1024 blocks x 8 waves
    gemm_layer1<<<(B_ * (O1_ / 32) * (N1_ / 64)) / 8, 256, 0, stream>>>(W1q, bias1, X, H1);

    // 5) MLP layer 2 -> d_out (B, 256, N1) f32
    gemm_layer2<<<(B_ * (O2_ / 32) * (N1_ / 64)) / 8, 256, 0, stream>>>(W2q, bias2, H1, out);
}
